// PureTriXStagedFFT_23081154249448
// MI455X (gfx1250) — matrix operand order repaired and checked by reading, not compiled, not run on hardware
//
#include <hip/hip_runtime.h>
#include <hip/hip_bf16.h>
#include <math.h>

// ---------------------------------------------------------------------------
// Types for CDNA5 WMMA (wave32): v_wmma_f32_16x16x32_bf16
// ---------------------------------------------------------------------------
typedef __attribute__((ext_vector_type(16))) __bf16 v16bf;
typedef __attribute__((ext_vector_type(8)))  __bf16 v8bf;
typedef __attribute__((ext_vector_type(8)))  float  v8f;
typedef __attribute__((ext_vector_type(4)))  unsigned v4u;

#define B_N    65536
#define D_D    512
#define T_T    8
#define CAPQ   131072   // max requests per expert bucket (2*B worst case)

// Workspace layout (bytes, all 256-aligned)
#define OFF_WIN   ((size_t)0)         // 288x512 bf16 frags   (294912 B)
#define OFF_WRS1  ((size_t)294912)    // 512x512              (524288 B)
#define OFF_WRS2  ((size_t)819200)    // 512x16               (16384 B)
#define OFF_WRD1  ((size_t)835584)    // 512x512              (524288 B)
#define OFF_WRD2  ((size_t)1359872)   // 512x16               (16384 B)
#define OFF_WT1   ((size_t)1376256)   // 8 x 512x1024         (8388608 B)
#define OFF_WT2   ((size_t)9764864)   // 8 x 1024x512         (8388608 B)
#define OFF_X     ((size_t)18153472)  // B x 512 bf16         (67108864 B)
#define OFF_CNT   ((size_t)85262336)  // 8 ints (+pad)        (256 B)
#define OFF_BKT   ((size_t)85262592)  // 8 x CAPQ ints        (4194304 B)

static __device__ __forceinline__ __bf16 f2bf(float f) {
  union { float f; unsigned u; } v; v.f = f;
  unsigned r = v.u + 0x7FFFu + ((v.u >> 16) & 1u);   // round-to-nearest-even
  unsigned short h = (unsigned short)(r >> 16);
  union { unsigned short u; __bf16 b; } o; o.u = h; return o.b;
}

static __device__ __forceinline__ float gelu_f(float x) {
  return 0.5f * x * (1.0f + erff(x * 0.70710678118654752440f));
}

// A-operand (16x32 bf16) fragment from row-major LDS tile.
// Contiguous-k layout (same as packed B): lane L (<16): row L, K = {k0..k0+15}
//             lane L+16   : row L, K = {k0+16..k0+31}
static __device__ __forceinline__ v16bf load_a_lds(const __bf16* base, int stride,
                                                   int k0, int lane) {
  int r = lane & 15;
  int koff = k0 + ((lane >> 4) << 4);
  const __bf16* p = base + r * stride + koff;
  v8bf lo = *(const v8bf*)p;
  v8bf hi = *(const v8bf*)(p + 8);
  return __builtin_shufflevector(lo, hi, 0,1,2,3,4,5,6,7,8,9,10,11,12,13,14,15);
}

#define WMMA_BF16(acc, A, Bf)                                                  \
  (acc) = __builtin_amdgcn_wmma_f32_16x16x32_bf16(false, (A), false, (Bf),     \
                                                  (short)0, (acc), false, false)

#define V8F_ZERO {0.f,0.f,0.f,0.f,0.f,0.f,0.f,0.f}

// ---------------------------------------------------------------------------
// Kernel 0: repack weights fp32 -> bf16 WMMA fragments; zero bucket counters.
// Fragment f = nb*(Kpad/32)+kb, 512 bf16 each, element (lane,i):
//   k = kb*32 + (lane>>4)*16 + i, n = nb*16 + (lane&15)
// so a lane's B operand is 16 contiguous bf16 (32B) at f*512 + lane*16.
// ---------------------------------------------------------------------------
static __device__ void pack_region(__bf16* dst, const float* src, int K, int Kpad,
                                   int Ncols, int Npad, long gid, long gsz) {
  long nkb = Kpad >> 5;
  long nnb = Npad >> 4;
  long total = nkb * nnb * 512;
  for (long e = gid; e < total; e += gsz) {
    long f  = e >> 9;
    int idx = (int)(e & 511);
    int kb  = (int)(f % nkb);
    int nb  = (int)(f / nkb);
    int lane = idx >> 4, i = idx & 15;
    int k = (kb << 5) + ((lane >> 4) << 4) + i;
    int n = (nb << 4) + (lane & 15);
    float v = (k < K && n < Ncols) ? src[(long)k * Ncols + n] : 0.0f;
    dst[e] = f2bf(v);
  }
}

__global__ void prep_kernel(const float* in_w, const float* rs_w1, const float* rs_w2,
                            const float* rd_w1, const float* rd_w2,
                            const float* tile_w1, const float* tile_w2, char* ws) {
  long gid = (long)blockIdx.x * blockDim.x + threadIdx.x;
  long gsz = (long)gridDim.x * blockDim.x;
  if (gid < 8) ((int*)(ws + OFF_CNT))[gid] = 0;
  pack_region((__bf16*)(ws + OFF_WIN),  in_w,  280, 288, 512, 512, gid, gsz);
  pack_region((__bf16*)(ws + OFF_WRS1), rs_w1, 512, 512, 512, 512, gid, gsz);
  pack_region((__bf16*)(ws + OFF_WRS2), rs_w2, 512, 512, 8,   16,  gid, gsz);
  pack_region((__bf16*)(ws + OFF_WRD1), rd_w1, 512, 512, 512, 512, gid, gsz);
  pack_region((__bf16*)(ws + OFF_WRD2), rd_w2, 512, 512, 8,   16,  gid, gsz);
  for (int t = 0; t < T_T; ++t) {
    pack_region((__bf16*)(ws + OFF_WT1) + (size_t)t * 524288,
                tile_w1 + (size_t)t * 512 * 1024, 512, 512, 1024, 1024, gid, gsz);
    pack_region((__bf16*)(ws + OFF_WT2) + (size_t)t * 524288,
                tile_w2 + (size_t)t * 1024 * 512, 1024, 1024, 512, 512, gid, gsz);
  }
}

// ---------------------------------------------------------------------------
// Kernel 1: featurize + LN + GELU + routers + bucket by expert.
// Block = 128 threads (4 waves), M = 64 rows. Waves own PAIRS of N-tiles:
// per kb step, 4 A fragments feed 8 WMMAs (2 B fragments) => 1 ds_load per
// WMMA and 4x weight reuse. Dynamic LDS layout (bytes):
//   [0,36864)         feat  64x288 bf16
//   [36864,167936)    y     64x512 f32
//   [167936,233472)   xb    64x512 bf16
//   [233472,299008)   hb    64x512 bf16
//   [299008,303104)   lg    64x16  f32
//   [303104,303616)   muL[64], rsL[64]
// ---------------------------------------------------------------------------
__global__ __launch_bounds__(128) void featroute_kernel(
    const int* stage, const int* pos, const float* aIn, const float* bIn,
    const float* stage_emb, const float* pos_emb,
    const float* in_b, const float* ln_g, const float* ln_b,
    const float* rs_b1, const float* rs_b2, const float* rd_b1, const float* rd_b2,
    char* ws) {
  extern __shared__ char smem[];
  __bf16* feat = (__bf16*)smem;
  float*  y    = (float*)(smem + 36864);
  __bf16* xb   = (__bf16*)(smem + 167936);
  __bf16* hb   = (__bf16*)(smem + 233472);
  float*  lg   = (float*)(smem + 299008);
  float*  muL  = (float*)(smem + 303104);
  float*  rsL  = muL + 64;

  const int tid = threadIdx.x;
  const int w = tid >> 5, lane = tid & 31;
  const int grow0 = blockIdx.x * 64;
  const int rb = (lane >> 4) * 8;

  // ---- build feature tile (64 x 288, zero-padded 280->288) ----
  for (int e = tid; e < 64 * 288; e += 128) {
    int r = e / 288, c = e % 288;
    int gr = grow0 + r;
    float v;
    if (c < 128)       v = stage_emb[stage[gr] * 128 + c];
    else if (c < 256)  v = pos_emb[pos[gr] * 128 + (c - 128)];
    else if (c < 280) {
      int j  = c - 256;
      float s = (j < 12) ? aIn[gr] : bIn[gr];
      int jj = j % 12;
      float ang = s * 0.0245436926061702596f * (float)(1 << (jj % 6)); // 2pi/256 * 2^k
      v = (jj < 6) ? sinf(ang) : cosf(ang);
    } else v = 0.0f;
    feat[r * 288 + c] = f2bf(v);
  }
  __syncthreads();

  // ---- input GEMM: (64x288) @ (288x512) ----
  {
    const __bf16* WIN = (const __bf16*)(ws + OFF_WIN);
    for (int nb0 = w * 2; nb0 < 32; nb0 += 8) {
      v8f acc0[4] = {V8F_ZERO, V8F_ZERO, V8F_ZERO, V8F_ZERO};
      v8f acc1[4] = {V8F_ZERO, V8F_ZERO, V8F_ZERO, V8F_ZERO};
      const __bf16* WB0 = WIN + (size_t)(nb0 * 9) * 512;
      const __bf16* WB1 = WB0 + (size_t)9 * 512;
      for (int kb = 0; kb < 9; ++kb) {
        v16bf B0 = *(const v16bf*)(WB0 + kb * 512 + lane * 16);
        v16bf B1 = *(const v16bf*)(WB1 + kb * 512 + lane * 16);
        #pragma unroll
        for (int m = 0; m < 4; ++m) {
          v16bf A = load_a_lds(feat + (m << 4) * 288, 288, kb * 32, lane);
          WMMA_BF16(acc0[m], A, B0);
          WMMA_BF16(acc1[m], A, B1);
        }
      }
      #pragma unroll
      for (int p = 0; p < 2; ++p) {
        int col = (nb0 + p) * 16 + (lane & 15);
        float bv = in_b[col];
        #pragma unroll
        for (int m = 0; m < 4; ++m)
          #pragma unroll
          for (int j = 0; j < 8; ++j)
            y[((m << 4) + rb + j) * 512 + col] = (p ? acc1[m][j] : acc0[m][j]) + bv;
      }
    }
  }
  __syncthreads();

  // ---- LayerNorm statistics ----
  if (tid < 64) {
    const float* yr = y + tid * 512;
    float s = 0.f, ss = 0.f;
    for (int i = 0; i < 512; ++i) { float v = yr[i]; s += v; ss += v * v; }
    float m = s * (1.0f / 512.0f);
    float var = ss * (1.0f / 512.0f) - m * m;
    muL[tid] = m;
    rsL[tid] = rsqrtf(var + 1e-5f);
  }
  __syncthreads();

  // ---- apply LN + GELU -> xb (LDS) and X (global, bf16) ----
  {
    __bf16* Xg = (__bf16*)(ws + OFF_X);
    for (int e = tid; e < 64 * 512; e += 128) {
      int r = e >> 9, c = e & 511;
      float v = (y[e] - muL[r]) * rsL[r] * ln_g[c] + ln_b[c];
      __bf16 h = f2bf(gelu_f(v));
      xb[e] = h;
      Xg[(size_t)(grow0 + r) * 512 + c] = h;
    }
  }
  __syncthreads();

  int* cnt = (int*)(ws + OFF_CNT);
  int* bkt = (int*)(ws + OFF_BKT);

  // ===================== two routers: SUM (which=0), DIFF (which=1) =========
  for (int which = 0; which < 2; ++which) {
    const __bf16* W1 = (const __bf16*)(ws + (which ? OFF_WRD1 : OFF_WRS1));
    const __bf16* W2 = (const __bf16*)(ws + (which ? OFF_WRD2 : OFF_WRS2));
    const float*  b1 = which ? rd_b1 : rs_b1;
    const float*  b2 = which ? rd_b2 : rs_b2;

    // h = gelu(x @ w1 + b1): (64x512) @ (512x512)
    for (int nb0 = w * 2; nb0 < 32; nb0 += 8) {
      v8f acc0[4] = {V8F_ZERO, V8F_ZERO, V8F_ZERO, V8F_ZERO};
      v8f acc1[4] = {V8F_ZERO, V8F_ZERO, V8F_ZERO, V8F_ZERO};
      const __bf16* WB0 = W1 + (size_t)(nb0 * 16) * 512;
      const __bf16* WB1 = WB0 + (size_t)16 * 512;
      for (int kb = 0; kb < 16; ++kb) {
        v16bf B0 = *(const v16bf*)(WB0 + kb * 512 + lane * 16);
        v16bf B1 = *(const v16bf*)(WB1 + kb * 512 + lane * 16);
        if (kb < 15) {
          __builtin_prefetch(WB0 + (kb + 1) * 512 + lane * 16, 0, 0);
          __builtin_prefetch(WB1 + (kb + 1) * 512 + lane * 16, 0, 0);
        }
        #pragma unroll
        for (int m = 0; m < 4; ++m) {
          v16bf A = load_a_lds(xb + (m << 4) * 512, 512, kb * 32, lane);
          WMMA_BF16(acc0[m], A, B0);
          WMMA_BF16(acc1[m], A, B1);
        }
      }
      #pragma unroll
      for (int p = 0; p < 2; ++p) {
        int col = (nb0 + p) * 16 + (lane & 15);
        float bv = b1[col];
        #pragma unroll
        for (int m = 0; m < 4; ++m)
          #pragma unroll
          for (int j = 0; j < 8; ++j)
            hb[((m << 4) + rb + j) * 512 + col] =
                f2bf(gelu_f((p ? acc1[m][j] : acc0[m][j]) + bv));
      }
    }
    __syncthreads();

    // logits = h @ w2 + b2: each wave computes its own 16-row tile (m = w)
    {
      v8f acc = V8F_ZERO;
      const __bf16* Abase = hb + (w << 4) * 512;
      for (int kb = 0; kb < 16; ++kb) {
        v16bf A  = load_a_lds(Abase, 512, kb * 32, lane);
        v16bf Bf = *(const v16bf*)(W2 + kb * 512 + lane * 16);
        WMMA_BF16(acc, A, Bf);
      }
      int n = lane & 15;
      float bv = (n < 8) ? b2[n] : 0.0f;
      #pragma unroll
      for (int j = 0; j < 8; ++j) lg[((w << 4) + rb + j) * 16 + n] = acc[j] + bv;
    }
    __syncthreads();

    if (tid < 64) {  // argmax (first-max) + bucket push
      const float* lr = lg + tid * 16;
      float best = lr[0]; int bi = 0;
      for (int c = 1; c < 8; ++c) { float v = lr[c]; if (v > best) { best = v; bi = c; } }
      int p = atomicAdd(&cnt[bi], 1);
      bkt[bi * CAPQ + p] = (((grow0 + tid) << 1) | which);
    }
    __syncthreads();   // lg / hb reused next iteration
  }
}

// ---------------------------------------------------------------------------
// Kernel 2: per-expert grouped GEMM. grid = (CAPQ/64, 8); blocks past an
// expert's count exit immediately. Block = 128 threads (4 waves) on 64
// gathered rows; paired N-tiles (1 ds_load per WMMA, 4x B reuse). Final
// 512-dot with sum_w/diff_w fused into the GEMM2 epilogue.
// LDS:
//   [0,256)           reqs[64]
//   [256,65792)       xr 64x512 bf16
//   [65792,196864)    hb 64x1024 bf16
//   [196864,197120)   pred[64] f32
// ---------------------------------------------------------------------------
__global__ __launch_bounds__(128) void expert_kernel(
    const float* tile_b1, const float* tile_b2,
    const float* sum_w, const float* sum_b, const float* diff_w, const float* diff_b,
    char* ws, float* out) {
  const int t = blockIdx.y;
  const int c = ((const int*)(ws + OFF_CNT))[t];
  const int base = blockIdx.x * 64;
  if (base >= c) return;

  extern __shared__ char smem[];
  int*    reqs = (int*)smem;
  __bf16* xr   = (__bf16*)(smem + 256);
  __bf16* hb   = (__bf16*)(smem + 65792);
  float*  pred = (float*)(smem + 196864);

  const int tid = threadIdx.x;
  const int w = tid >> 5, lane = tid & 31;
  const int rb = (lane >> 4) * 8;
  const int* bkt = (const int*)(ws + OFF_BKT) + (size_t)t * CAPQ;

  if (tid < 64) {
    int idx = base + tid;
    if (idx >= c) idx = base;       // pad chunk with a valid request (not written)
    reqs[tid] = bkt[idx];
    pred[tid] = 0.0f;
  }
  __syncthreads();

  // gather 64 rows of X (bf16) with 16-byte copies
  {
    const v4u* Xg = (const v4u*)(ws + OFF_X);
    v4u* xr16 = (v4u*)xr;
    for (int e = tid; e < 64 * 64; e += 128) {
      int r = e >> 6, c2 = e & 63;
      int row = reqs[r] >> 1;
      xr16[(r << 6) + c2] = Xg[(size_t)row * 64 + c2];
    }
  }
  __syncthreads();
  // per-lane sum/diff selection mask: bit (m*8+j) = which of local row m*16+rb+j
  unsigned whMask = 0;
  #pragma unroll
  for (int m = 0; m < 4; ++m)
    #pragma unroll
    for (int j = 0; j < 8; ++j)
      whMask |= (unsigned)(reqs[(m << 4) + rb + j] & 1) << (m * 8 + j);

  // h = gelu(xr @ w1_t + b1_t)  : 64x512 -> 64x1024
  {
    const __bf16* W1 = (const __bf16*)(ws + OFF_WT1) + (size_t)t * 524288;
    for (int nb0 = w * 2; nb0 < 64; nb0 += 8) {
      v8f acc0[4] = {V8F_ZERO, V8F_ZERO, V8F_ZERO, V8F_ZERO};
      v8f acc1[4] = {V8F_ZERO, V8F_ZERO, V8F_ZERO, V8F_ZERO};
      const __bf16* WB0 = W1 + (size_t)(nb0 * 16) * 512;
      const __bf16* WB1 = WB0 + (size_t)16 * 512;
      for (int kb = 0; kb < 16; ++kb) {
        v16bf B0 = *(const v16bf*)(WB0 + kb * 512 + lane * 16);
        v16bf B1 = *(const v16bf*)(WB1 + kb * 512 + lane * 16);
        if (kb < 15) {
          __builtin_prefetch(WB0 + (kb + 1) * 512 + lane * 16, 0, 0);
          __builtin_prefetch(WB1 + (kb + 1) * 512 + lane * 16, 0, 0);
        }
        #pragma unroll
        for (int m = 0; m < 4; ++m) {
          v16bf A = load_a_lds(xr + (m << 4) * 512, 512, kb * 32, lane);
          WMMA_BF16(acc0[m], A, B0);
          WMMA_BF16(acc1[m], A, B1);
        }
      }
      #pragma unroll
      for (int p = 0; p < 2; ++p) {
        int col = (nb0 + p) * 16 + (lane & 15);
        float bv = tile_b1[t * 1024 + col];
        #pragma unroll
        for (int m = 0; m < 4; ++m)
          #pragma unroll
          for (int j = 0; j < 8; ++j)
            hb[((m << 4) + rb + j) * 1024 + col] =
                f2bf(gelu_f((p ? acc1[m][j] : acc0[m][j]) + bv));
      }
    }
  }
  __syncthreads();

  // o = h @ w2_t + b2_t (64x1024 -> 64x512), fused with final dot:
  // pred[row] += sum_col (o[row][col] * (which ? diff_w : sum_w)[col])
  {
    const __bf16* W2 = (const __bf16*)(ws + OFF_WT2) + (size_t)t * 524288;
    float part[4][8];
    #pragma unroll
    for (int m = 0; m < 4; ++m)
      #pragma unroll
      for (int j = 0; j < 8; ++j) part[m][j] = 0.0f;

    for (int nb0 = w * 2; nb0 < 32; nb0 += 8) {
      v8f acc0[4] = {V8F_ZERO, V8F_ZERO, V8F_ZERO, V8F_ZERO};
      v8f acc1[4] = {V8F_ZERO, V8F_ZERO, V8F_ZERO, V8F_ZERO};
      const __bf16* WB0 = W2 + (size_t)(nb0 * 32) * 512;
      const __bf16* WB1 = WB0 + (size_t)32 * 512;
      for (int kb = 0; kb < 32; ++kb) {
        v16bf B0 = *(const v16bf*)(WB0 + kb * 512 + lane * 16);
        v16bf B1 = *(const v16bf*)(WB1 + kb * 512 + lane * 16);
        if (kb < 31) {
          __builtin_prefetch(WB0 + (kb + 1) * 512 + lane * 16, 0, 0);
          __builtin_prefetch(WB1 + (kb + 1) * 512 + lane * 16, 0, 0);
        }
        #pragma unroll
        for (int m = 0; m < 4; ++m) {
          v16bf A = load_a_lds(hb + (m << 4) * 1024, 1024, kb * 32, lane);
          WMMA_BF16(acc0[m], A, B0);
          WMMA_BF16(acc1[m], A, B1);
        }
      }
      #pragma unroll
      for (int p = 0; p < 2; ++p) {
        int col = (nb0 + p) * 16 + (lane & 15);
        float b2c = tile_b2[t * 512 + col];
        float wsc = sum_w[col], wdc = diff_w[col];
        #pragma unroll
        for (int m = 0; m < 4; ++m)
          #pragma unroll
          for (int j = 0; j < 8; ++j) {
            float o  = (p ? acc1[m][j] : acc0[m][j]) + b2c;
            float wv = ((whMask >> (m * 8 + j)) & 1u) ? wdc : wsc;
            part[m][j] += o * wv;
          }
      }
    }
    #pragma unroll
    for (int m = 0; m < 4; ++m)
      #pragma unroll
      for (int j = 0; j < 8; ++j)
        atomicAdd(&pred[(m << 4) + rb + j], part[m][j]);
  }
  __syncthreads();

  // scatter to output
  if (tid < 64 && base + tid < c) {
    int req = reqs[tid];
    int row = req >> 1, which = req & 1;
    float bb = which ? diff_b[0] : sum_b[0];
    out[(size_t)which * B_N + row] = pred[tid] + bb;
  }
}

// ---------------------------------------------------------------------------
extern "C" void kernel_launch(void* const* d_in, const int* in_sizes, int n_in,
                              void* d_out, int out_size, void* d_ws, size_t ws_size,
                              hipStream_t stream) {
  (void)in_sizes; (void)n_in; (void)out_size; (void)ws_size;
  char* ws = (char*)d_ws;

  const int*   stage     = (const int*)  d_in[0];
  const int*   pos       = (const int*)  d_in[1];
  const float* a         = (const float*)d_in[2];
  const float* b         = (const float*)d_in[3];
  const float* stage_emb = (const float*)d_in[4];
  const float* pos_emb   = (const float*)d_in[5];
  const float* in_w      = (const float*)d_in[6];
  const float* in_b      = (const float*)d_in[7];
  const float* ln_g      = (const float*)d_in[8];
  const float* ln_b      = (const float*)d_in[9];
  const float* rs_w1     = (const float*)d_in[10];
  const float* rs_b1     = (const float*)d_in[11];
  const float* rs_w2     = (const float*)d_in[12];
  const float* rs_b2     = (const float*)d_in[13];
  const float* rd_w1     = (const float*)d_in[14];
  const float* rd_b1     = (const float*)d_in[15];
  const float* rd_w2     = (const float*)d_in[16];
  const float* rd_b2     = (const float*)d_in[17];
  const float* tile_w1   = (const float*)d_in[18];
  const float* tile_b1   = (const float*)d_in[19];
  const float* tile_w2   = (const float*)d_in[20];
  const float* tile_b2   = (const float*)d_in[21];
  const float* sum_w     = (const float*)d_in[22];
  const float* sum_b     = (const float*)d_in[23];
  const float* diff_w    = (const float*)d_in[24];
  const float* diff_b    = (const float*)d_in[25];
  float* out = (float*)d_out;

  (void)hipFuncSetAttribute((const void*)featroute_kernel,
                            hipFuncAttributeMaxDynamicSharedMemorySize, 303616);
  (void)hipFuncSetAttribute((const void*)expert_kernel,
                            hipFuncAttributeMaxDynamicSharedMemorySize, 197120);

  prep_kernel<<<1024, 256, 0, stream>>>(in_w, rs_w1, rs_w2, rd_w1, rd_w2,
                                        tile_w1, tile_w2, ws);

  featroute_kernel<<<B_N / 64, 128, 303616, stream>>>(
      stage, pos, a, b, stage_emb, pos_emb, in_b, ln_g, ln_b,
      rs_b1, rs_b2, rd_b1, rd_b2, ws);

  dim3 g2(CAPQ / 64, T_T);
  expert_kernel<<<g2, 128, 197120, stream>>>(tile_b1, tile_b2, sum_w, sum_b,
                                             diff_w, diff_b, ws, out);
}
